// SpaceTimeHashingField_27848567947936
// MI455X (gfx1250) — matrix operand order repaired and checked by reading, not compiled, not run on hardware
//
#include <hip/hip_runtime.h>
#include <stdint.h>
#include <stddef.h>

typedef __attribute__((ext_vector_type(16))) __bf16 v16bf;
typedef __attribute__((ext_vector_type(8)))  __bf16 v8bf;
typedef __attribute__((ext_vector_type(8)))  float  v8f;

#define NPTS 300000
#define TBL  (1u << 19)
#define BLK  256
#define FRAG_ELEMS 512   // 32 lanes x 16 bf16 per B-fragment

#define WMMA_BF16(A, B, C) \
  __builtin_amdgcn_wmma_f32_16x16x32_bf16(false, (A), false, (B), (short)0, (C), false, false)

// Single-instruction ReLU: v_med3_f32(x, 0, +MAX) (no canonicalize needed)
__device__ __forceinline__ float relu(float x) {
  return __builtin_amdgcn_fmed3f(x, 0.f, 3.402823466e+38f);
}

// ---- A fragment: 16(M) x 32(K) bf16 from LDS (ISA 7.12.2 layout) ----
__device__ __forceinline__ v16bf load_a(const __bf16* lds, int stride,
                                        int row0, int kbase, int lane) {
  const __bf16* p = lds + (row0 + (lane & 15)) * stride + kbase + ((lane >> 4) << 3);
  v8bf lo = *reinterpret_cast<const v8bf*>(p);        // ds_load_b128
  v8bf hi = *reinterpret_cast<const v8bf*>(p + 16);   // ds_load_b128
  return __builtin_shufflevector(lo, hi, 0, 1, 2, 3, 4, 5, 6, 7,
                                 8, 9, 10, 11, 12, 13, 14, 15);
}

// ---- B fragment: pre-packed bf16, 32 contiguous bytes per lane ----
__device__ __forceinline__ v16bf load_bp(const __bf16* __restrict__ P,
                                         int frag, int lane) {
  return *reinterpret_cast<const v16bf*>(P + (size_t)frag * FRAG_ELEMS + lane * 16);
}

// ---- [KC*32 x 64] layer + ReLU for both 16-point tiles, A-frags given ----
template <int KC>
__device__ __forceinline__ void layer_frag(const v16bf a0[KC], const v16bf a1[KC],
                                           const __bf16* __restrict__ P, int fragBase,
                                           __bf16* outp, int tb0, int lane) {
  int m0 = (lane >> 4) << 3;
  int nl = lane & 15;
#pragma unroll
  for (int nt = 0; nt < 4; ++nt) {
    v8f acc0 = {}, acc1 = {};
#pragma unroll
    for (int kc = 0; kc < KC; ++kc) {
      v16bf b = load_bp(P, fragBase + kc * 4 + nt, lane);
      acc0 = WMMA_BF16(a0[kc], b, acc0);
      acc1 = WMMA_BF16(a1[kc], b, acc1);
    }
    int cbase = (nt << 4) + nl;
#pragma unroll
    for (int r = 0; r < 8; ++r) {
      outp[(tb0 + m0 + r) * 64 + cbase]      = (__bf16)relu(acc0[r]);
      outp[(tb0 + 16 + m0 + r) * 64 + cbase] = (__bf16)relu(acc1[r]);
    }
  }
}

template <int KC>
__device__ __forceinline__ void load_a2(const __bf16* in, int tb0, int lane,
                                        v16bf a0[KC], v16bf a1[KC]) {
#pragma unroll
  for (int kc = 0; kc < KC; ++kc) {
    a0[kc] = load_a(in, KC * 32, tb0,      kc << 5, lane);
    a1[kc] = load_a(in, KC * 32, tb0 + 16, kc << 5, lane);
  }
}

template <int KC>
__device__ __forceinline__ void layer64x2(const __bf16* in, const __bf16* __restrict__ P,
                                          int fragBase, __bf16* outp, int tb0, int lane) {
  v16bf a0[KC], a1[KC];
  load_a2<KC>(in, tb0, lane, a0, a1);
  layer_frag<KC>(a0, a1, P, fragBase, outp, tb0, lane);
}

// ---- Final [64 x OD] layer, fused residual + mask, NT scatter to global ----
template <int NTILES, int OD>
__device__ __forceinline__ void head_final2(const __bf16* in, const __bf16* __restrict__ P,
                                            int fragBase,
                                            const float* __restrict__ residual,
                                            float* __restrict__ outp,
                                            v8f mk0, v8f mk1,
                                            int tb0, int blockBase, int lane) {
  int m0 = (lane >> 4) << 3;
  int nl = lane & 15;
  v16bf a0[2], a1[2];
  load_a2<2>(in, tb0, lane, a0, a1);
#pragma unroll
  for (int nt = 0; nt < NTILES; ++nt) {
    v8f acc0 = {}, acc1 = {};
#pragma unroll
    for (int kc = 0; kc < 2; ++kc) {
      v16bf b = load_bp(P, fragBase + kc * NTILES + nt, lane);
      acc0 = WMMA_BF16(a0[kc], b, acc0);
      acc1 = WMMA_BF16(a1[kc], b, acc1);
    }
    int n = (nt << 4) + nl;
    if (n < OD) {
#pragma unroll
      for (int r = 0; r < 8; ++r) {
        int pl0 = tb0 + m0 + r;
        int p0  = blockBase + pl0;
        int p1  = p0 + 16;
        if (p0 < NPTS) {
          size_t ix = (size_t)p0 * OD + n;
          float v = __builtin_nontemporal_load(&residual[ix]) + mk0[r] * acc0[r];
          __builtin_nontemporal_store(v, &outp[ix]);
        }
        if (p1 < NPTS) {
          size_t ix = (size_t)p1 * OD + n;
          float v = __builtin_nontemporal_load(&residual[ix]) + mk1[r] * acc1[r];
          __builtin_nontemporal_store(v, &outp[ix]);
        }
      }
    }
  }
}

// ---- One-time weight pack: f32 row-major [K][ncols] -> B-fragment bf16 ----
__global__ void pack_weights_kernel(const float* __restrict__ W, int ncols, int ntiles,
                                    __bf16* __restrict__ P) {
  int frag = blockIdx.x;                 // frag = kc * ntiles + nt
  int kc = frag / ntiles;
  int nt = frag - kc * ntiles;
  int lane = threadIdx.x;
  int n  = (lane & 15) + (nt << 4);
  int kb = (kc << 5) + ((lane >> 4) << 3);
  bool valid = n < ncols;
  const float* Wp = W + (size_t)kb * ncols + (valid ? n : 0);
  v16bf b;
#pragma unroll
  for (int h = 0; h < 16; ++h)
    b[h] = valid ? (__bf16)Wp[(size_t)((h & 7) + ((h >> 3) << 4)) * ncols] : (__bf16)0.f;
  *reinterpret_cast<v16bf*>(P + (size_t)frag * FRAG_ELEMS + lane * 16) = b;
}

__global__ __launch_bounds__(BLK)
void spacetime_hash_field_kernel(
    const float* __restrict__ xyz,   const float* __restrict__ scales,
    const float* __restrict__ rot,   const float* __restrict__ opac,
    const float* __restrict__ shs,   const float* __restrict__ timep,
    const float* __restrict__ bmin,  const float* __restrict__ bmax,
    const float* __restrict__ table, const __bf16* __restrict__ Wp,
    float* __restrict__ out) {
  __shared__ __bf16 encL[BLK * 32];
  __shared__ __bf16 actA[BLK * 64];
  __shared__ __bf16 actB[BLK * 64];
  __shared__ __bf16 hid[BLK * 64];
  __shared__ __align__(32) float maskL[BLK];

  const int tid = threadIdx.x;
  const int blockBase = blockIdx.x * BLK;
  int p  = blockBase + tid;
  int pc = p < NPTS ? p : NPTS - 1;

  // ---------------- phase 1: 4-D multiresolution hash encoding ------------
  float x4[4];
  bool ok = true;
#pragma unroll
  for (int c = 0; c < 3; ++c) {
    float v = (__builtin_nontemporal_load(&xyz[(size_t)pc * 3 + c]) - bmin[c]) /
              (bmax[c] - bmin[c]);
    ok = ok && (v >= 0.f) && (v <= 1.f);
    x4[c] = v;
  }
  x4[3] = __builtin_nontemporal_load(&timep[pc]);
  maskL[tid] = ok ? 1.f : 0.f;
#pragma unroll
  for (int c = 0; c < 4; ++c) x4[c] = fminf(fmaxf(x4[c], 0.f), 1.f);

  const unsigned int PR[4] = {1u, 2654435761u, 805459861u, 3674653429u};
  for (int l = 0; l < 16; ++l) {
    float s = (float)(16u << l) - 1.f;
    unsigned int term[4][2];
    float wsel[4][2];
#pragma unroll
    for (int c = 0; c < 4; ++c) {
      float pos = fmaf(x4[c], s, 0.5f);
      float fl  = floorf(pos);
      float w   = pos - fl;
      unsigned int base = (unsigned int)(int)fl * PR[c];  // (p0+b)*PR = p0*PR + b*PR
      term[c][0] = base;
      term[c][1] = base + PR[c];
      wsel[c][0] = 1.f - w;
      wsel[c][1] = w;
    }
    float p01[4], p23[4];
#pragma unroll
    for (int c = 0; c < 4; ++c) {
      p01[c] = wsel[0][c & 1] * wsel[1][(c >> 1) & 1];
      p23[c] = wsel[2][c & 1] * wsel[3][(c >> 1) & 1];
    }
    const float* tl = table + (size_t)l * TBL * 2u;
    float a0 = 0.f, a1 = 0.f;
#pragma unroll
    for (int c = 0; c < 16; ++c) {
      unsigned int h = (term[0][c & 1] ^ term[1][(c >> 1) & 1] ^
                        term[2][(c >> 2) & 1] ^ term[3][(c >> 3) & 1]) & (TBL - 1u);
      float wc = p01[c & 3] * p23[(c >> 2) & 3];
      const float2 f2 = *reinterpret_cast<const float2*>(tl + 2u * h); // RT: stays L2-resident
      a0 = fmaf(f2.x, wc, a0);
      a1 = fmaf(f2.y, wc, a1);
    }
    encL[tid * 32 + 2 * l]     = (__bf16)a0;
    encL[tid * 32 + 2 * l + 1] = (__bf16)a1;
  }
  __syncthreads();

  // ---------------- phase 2: WMMA MLP trunk + 5 heads ---------------------
  const int wave = tid >> 5;
  const int lane = tid & 31;
  const int tb0  = wave * 32;
  const int m0   = (lane >> 4) << 3;
  const size_t Nn = NPTS;

  layer64x2<1>(encL, Wp, 0, actA, tb0, lane);   // enc(32) -> 64, ReLU
  layer64x2<2>(actA, Wp, 4, hid,  tb0, lane);   // 64 -> 64, ReLU (trunk)

  // Trunk A-fragments + per-point masks: load once, reuse across all 5 heads.
  v16bf h0[2], h1[2];
  load_a2<2>(hid, tb0, lane, h0, h1);
  v8f mk0 = *reinterpret_cast<const v8f*>(&maskL[tb0 + m0]);
  v8f mk1 = *reinterpret_cast<const v8f*>(&maskL[tb0 + 16 + m0]);

  // xyz head (frags 12/20/28)
  layer_frag<2>(h0, h1, Wp, 12, actA, tb0, lane);
  layer64x2<2>(actA, Wp, 20, actB, tb0, lane);
  head_final2<1, 3>(actB, Wp, 28, xyz, out, mk0, mk1, tb0, blockBase, lane);
  // scales head (30/38/46)
  layer_frag<2>(h0, h1, Wp, 30, actA, tb0, lane);
  layer64x2<2>(actA, Wp, 38, actB, tb0, lane);
  head_final2<1, 3>(actB, Wp, 46, scales, out + 3 * Nn, mk0, mk1, tb0, blockBase, lane);
  // rotation head (48/56/64)
  layer_frag<2>(h0, h1, Wp, 48, actA, tb0, lane);
  layer64x2<2>(actA, Wp, 56, actB, tb0, lane);
  head_final2<1, 4>(actB, Wp, 64, rot, out + 6 * Nn, mk0, mk1, tb0, blockBase, lane);
  // opacity head (66/74/82)
  layer_frag<2>(h0, h1, Wp, 66, actA, tb0, lane);
  layer64x2<2>(actA, Wp, 74, actB, tb0, lane);
  head_final2<1, 1>(actB, Wp, 82, opac, out + 10 * Nn, mk0, mk1, tb0, blockBase, lane);
  // shs head (84/92/100)
  layer_frag<2>(h0, h1, Wp, 84, actA, tb0, lane);
  layer64x2<2>(actA, Wp, 92, actB, tb0, lane);
  head_final2<3, 48>(actB, Wp, 100, shs, out + 11 * Nn, mk0, mk1, tb0, blockBase, lane);
}

extern "C" void kernel_launch(void* const* d_in, const int* in_sizes, int n_in,
                              void* d_out, int out_size, void* d_ws, size_t ws_size,
                              hipStream_t stream) {
  (void)in_sizes; (void)n_in; (void)ws_size; (void)out_size;
  const float* xyz    = (const float*)d_in[0];
  const float* scales = (const float*)d_in[1];
  const float* rot    = (const float*)d_in[2];
  const float* opac   = (const float*)d_in[3];
  const float* shs    = (const float*)d_in[4];
  const float* timep  = (const float*)d_in[5];
  const float* bmin   = (const float*)d_in[6];
  const float* bmax   = (const float*)d_in[7];
  const float* table  = (const float*)d_in[8];

  __bf16* Wpk = (__bf16*)d_ws;   // 106 fragments * 1KB = ~106 KB scratch

  struct Desc { int idx, ncols, kchunks, ntiles, base; };
  const Desc descs[17] = {
      {9, 64, 1, 4, 0},    {10, 64, 2, 4, 4},                        // We0, We1
      {11, 64, 2, 4, 12},  {12, 64, 2, 4, 20},  {13, 3, 2, 1, 28},   // xyz head
      {14, 64, 2, 4, 30},  {15, 64, 2, 4, 38},  {16, 3, 2, 1, 46},   // scl head
      {17, 64, 2, 4, 48},  {18, 64, 2, 4, 56},  {19, 4, 2, 1, 64},   // rot head
      {20, 64, 2, 4, 66},  {21, 64, 2, 4, 74},  {22, 1, 2, 1, 82},   // op head
      {23, 64, 2, 4, 84},  {24, 64, 2, 4, 92},  {25, 48, 2, 3, 100}, // shs head
  };
  for (int i = 0; i < 17; ++i) {
    const Desc& d = descs[i];
    pack_weights_kernel<<<dim3(d.kchunks * d.ntiles), dim3(32), 0, stream>>>(
        (const float*)d_in[d.idx], d.ncols, d.ntiles,
        Wpk + (size_t)d.base * FRAG_ELEMS);
  }

  dim3 grid((NPTS + BLK - 1) / BLK), block(BLK);
  spacetime_hash_field_kernel<<<grid, block, 0, stream>>>(
      xyz, scales, rot, opac, shs, timep, bmin, bmax, table, Wpk, (float*)d_out);
}
